// ModulatedLinear_82652350644272
// MI455X (gfx1250) — matrix operand order repaired and checked by reading, not compiled, not run on hardware
//
#include <hip/hip_runtime.h>

typedef __attribute__((ext_vector_type(16))) _Float16 v16h;
typedef __attribute__((ext_vector_type(8)))  _Float16 v8h;
typedef __attribute__((ext_vector_type(8)))  float    v8f;
typedef __attribute__((ext_vector_type(4)))  float    v4f;
typedef __attribute__((ext_vector_type(4)))  int      v4i;

constexpr int kB   = 16;
constexpr int kN   = 4096;
constexpr int kIN  = 512;
constexpr int kOUT = 512;
constexpr int kWF  = 512;
constexpr float kEPS = 1e-8f;

constexpr int BM = 128;   // block tile M (rows of x)
constexpr int BN = 128;   // block tile N (cols = out)
constexpr int BK = 32;    // K step = WMMA K
constexpr int KSTEPS = kIN / BK;   // 16

// ---------------------------------------------------------------------------
// CDNA5 async global->LDS copy (16B) + asynccnt wait, with portable fallbacks.
// ---------------------------------------------------------------------------
__device__ __forceinline__ void async_cp16(const _Float16* g, _Float16* l) {
#if defined(__gfx1250__) && __has_builtin(__builtin_amdgcn_global_load_async_to_lds_b128)
    __builtin_amdgcn_global_load_async_to_lds_b128((v4i*)g, (v4i*)l, 0, 0);
#else
    *(v8h*)l = *(const v8h*)g;   // synchronous fallback (ds_store path)
#endif
}

__device__ __forceinline__ void wait_async0() {
#if defined(__gfx1250__) && __has_builtin(__builtin_amdgcn_global_load_async_to_lds_b128)
#if __has_builtin(__builtin_amdgcn_s_wait_asynccnt)
    __builtin_amdgcn_s_wait_asynccnt(0);
#else
    asm volatile("s_wait_asynccnt 0" ::: "memory");
#endif
#endif
}

// ---------------------------------------------------------------------------
// Kernel 1: s[b, i] = sum_k w[b, k] * A_w[i, k] + A_b[i]     (16 x 512, tiny)
// ---------------------------------------------------------------------------
__global__ __launch_bounds__(512)
void style_kernel(const float* __restrict__ w, const float* __restrict__ A_w,
                  const float* __restrict__ A_b, float* __restrict__ s) {
    __shared__ float wsm[kWF];
    const int b = blockIdx.x;
    const int t = threadIdx.x;
    wsm[t] = w[b * kWF + t];
    __syncthreads();
    const float* arow = A_w + (size_t)t * kWF;
    float acc = 0.f;
    #pragma unroll 4
    for (int k = 0; k < kWF; k += 4) {
        v4f a = *(const v4f*)(arow + k);
        acc += a.x * wsm[k] + a.y * wsm[k + 1] + a.z * wsm[k + 2] + a.w * wsm[k + 3];
    }
    s[b * kIN + t] = acc + A_b[t];
}

// ---------------------------------------------------------------------------
// Kernel 2: wd[b, o, i] = f16( weight[o,i]*s[b,i] * rsqrt(eps + sum_i (.)^2) )
// ---------------------------------------------------------------------------
__global__ __launch_bounds__(128)
void demod_kernel(const float* __restrict__ weight, const float* __restrict__ s,
                  _Float16* __restrict__ wd) {
    const int o = blockIdx.x, b = blockIdx.y, t = threadIdx.x;
    const float* wrow = weight + (size_t)o * kIN;
    const float* srow = s + (size_t)b * kIN;
    float wm[4];
    float ssq = 0.f;
    #pragma unroll
    for (int j = 0; j < 4; ++j) {
        const int i = t + j * 128;
        wm[j] = wrow[i] * srow[i];
        ssq += wm[j] * wm[j];
    }
    #pragma unroll
    for (int off = 16; off > 0; off >>= 1) ssq += __shfl_down(ssq, off, 32);
    __shared__ float red[4];
    if ((t & 31) == 0) red[t >> 5] = ssq;
    __syncthreads();
    const float norm = rsqrtf(kEPS + red[0] + red[1] + red[2] + red[3]);
    _Float16* drow = wd + ((size_t)b * kOUT + o) * kIN;
    #pragma unroll
    for (int j = 0; j < 4; ++j) {
        const int i = t + j * 128;
        drow[i] = (_Float16)(wm[j] * norm);
    }
}

// ---------------------------------------------------------------------------
// Kernel 3: out[b, n, o] = relu( x[b] @ wd[b].T + bias )
// 256 threads = 8 waves (4 M x 2 N), block tile 128x128, double-buffered LDS.
//   A tile: x 128x32 f32 -> f16, stored in WMMA A-fragment order
//           Al[stage][mtile(8)][half(2)][lane(32)][8]
//   B tile: wd 128 cols x 32 k f16, async global->LDS copy
//           Bl[stage][col(128)][k(32)]
// All 12 fragment ds_loads are issued before the 8 WMMAs of a K-step so the
// matrix pipe gets back-to-back issue after a single dscnt wait.
// ---------------------------------------------------------------------------
__global__ __launch_bounds__(256)
void mod_gemm_kernel(const float* __restrict__ x, const _Float16* __restrict__ wd,
                     const float* __restrict__ bias, float* __restrict__ out) {
    __shared__ _Float16 Al[2][8][2][32][8];   // 16 KB
    __shared__ _Float16 Bl[2][BN][BK];        // 16 KB

    const int b    = blockIdx.z;
    const int tid  = threadIdx.x;
    const int wave = tid >> 5;
    const int lane = tid & 31;
    const int lh   = lane & 15;
    const int hi   = lane >> 4;

    const int wmid = wave & 3;    // 4 waves along M
    const int wnid = wave >> 2;   // 2 waves along N
    const int m_blk = blockIdx.x * BM;
    const int n_blk = blockIdx.y * BN;

    const float*    xb = x  + (size_t)b * kN * kIN;
    const _Float16* wb = wd + (size_t)b * kOUT * kIN;

    // --- staging roles (per thread) ---
    const int ar = tid >> 1;           // A: row
    const int cg = (tid & 1) << 4;     // A: column-group (16 floats)
    const int atile = ar >> 4;
    const int alane = ar & 15;
    const int ahalf = cg >> 4;
    const float* apx = xb + (size_t)(m_blk + ar) * kIN + cg;
    const int bc0 = tid, bc1 = tid + 256;   // B: 16B chunk ids
    const _Float16* bp0 = wb + (size_t)(n_blk + (bc0 >> 2)) * kIN + ((bc0 & 3) << 3);
    const _Float16* bp1 = wb + (size_t)(n_blk + (bc1 >> 2)) * kIN + ((bc1 & 3) << 3);

    v8f acc[2][4] = {};

    // ---- prologue: fill stage 0 ----
    {
        v4f x0 = *(const v4f*)(apx);
        v4f x1 = *(const v4f*)(apx + 4);
        v4f x2 = *(const v4f*)(apx + 8);
        v4f x3 = *(const v4f*)(apx + 12);
        v8h hlo, hhi;
        #pragma unroll
        for (int j = 0; j < 4; ++j) {
            hlo[j]     = (_Float16)x0[j];
            hlo[4 + j] = (_Float16)x1[j];
            hhi[j]     = (_Float16)x2[j];
            hhi[4 + j] = (_Float16)x3[j];
        }
        *(v8h*)&Al[0][atile][ahalf][alane][0]      = hlo;
        *(v8h*)&Al[0][atile][ahalf][alane + 16][0] = hhi;
        async_cp16(bp0, &Bl[0][bc0 >> 2][(bc0 & 3) << 3]);
        async_cp16(bp1, &Bl[0][bc1 >> 2][(bc1 & 3) << 3]);
        wait_async0();
    }
    __syncthreads();

    for (int ks = 0; ks < KSTEPS; ++ks) {
        const int st  = ks & 1;
        const int stn = st ^ 1;
        const bool has_next = (ks + 1) < KSTEPS;
        const int kn = (ks + 1) * BK;

        // ---- prefetch next stage: A f32 -> regs, B async -> LDS ----
        v4f x0, x1, x2, x3;
        if (has_next) {
            x0 = *(const v4f*)(apx + kn);
            x1 = *(const v4f*)(apx + kn + 4);
            x2 = *(const v4f*)(apx + kn + 8);
            x3 = *(const v4f*)(apx + kn + 12);
            async_cp16(bp0 + kn, &Bl[stn][bc0 >> 2][(bc0 & 3) << 3]);
            async_cp16(bp1 + kn, &Bl[stn][bc1 >> 2][(bc1 & 3) << 3]);
        }

        // ---- load ALL fragments for this K-step, then 8 back-to-back WMMAs ----
        v16h a[2];
        v16h bf[4];
        #pragma unroll
        for (int mt = 0; mt < 2; ++mt) {
            const int tl = wmid * 2 + mt;
            v8h alo = *(const v8h*)&Al[st][tl][0][lane][0];
            v8h ahi = *(const v8h*)&Al[st][tl][1][lane][0];
            #pragma unroll
            for (int j = 0; j < 8; ++j) { a[mt][j] = alo[j]; a[mt][8 + j] = ahi[j]; }
        }
        #pragma unroll
        for (int nt = 0; nt < 4; ++nt) {
            const int col = wnid * 64 + nt * 16 + lh;
            v8h b0 = *(const v8h*)&Bl[st][col][hi * 8];
            v8h b1 = *(const v8h*)&Bl[st][col][16 + hi * 8];
            #pragma unroll
            for (int j = 0; j < 8; ++j) { bf[nt][j] = b0[j]; bf[nt][8 + j] = b1[j]; }
        }
        #pragma unroll
        for (int nt = 0; nt < 4; ++nt) {
            #pragma unroll
            for (int mt = 0; mt < 2; ++mt) {
                acc[mt][nt] = __builtin_amdgcn_wmma_f32_16x16x32_f16(
                    false, a[mt], false, bf[nt], (short)0, acc[mt][nt], false, false);
            }
        }

        // ---- stage A regs into next-stage LDS ----
        if (has_next) {
            v8h hlo, hhi;
            #pragma unroll
            for (int j = 0; j < 4; ++j) {
                hlo[j]     = (_Float16)x0[j];
                hlo[4 + j] = (_Float16)x1[j];
                hhi[j]     = (_Float16)x2[j];
                hhi[4 + j] = (_Float16)x3[j];
            }
            *(v8h*)&Al[stn][atile][ahalf][alane][0]      = hlo;
            *(v8h*)&Al[stn][atile][ahalf][alane + 16][0] = hhi;
            wait_async0();
        }
        __syncthreads();
    }

    // ---- epilogue: bias + relu, f32 store ----
    float* ob = out + (size_t)b * kN * kOUT;
    #pragma unroll
    for (int nt = 0; nt < 4; ++nt) {
        const int col = n_blk + wnid * 64 + nt * 16 + lh;
        const float bv = bias[col];
        #pragma unroll
        for (int mt = 0; mt < 2; ++mt) {
            #pragma unroll
            for (int r = 0; r < 8; ++r) {
                const int row = m_blk + wmid * 32 + mt * 16 + r + hi * 8;
                float v = acc[mt][nt][r] + bv;
                ob[(size_t)row * kOUT + col] = v > 0.f ? v : 0.f;
            }
        }
    }
}

// ---------------------------------------------------------------------------
extern "C" void kernel_launch(void* const* d_in, const int* in_sizes, int n_in,
                              void* d_out, int out_size, void* d_ws, size_t ws_size,
                              hipStream_t stream) {
    const float* x      = (const float*)d_in[0];
    const float* w      = (const float*)d_in[1];
    const float* weight = (const float*)d_in[2];
    const float* bias   = (const float*)d_in[3];
    const float* A_w    = (const float*)d_in[4];
    const float* A_b    = (const float*)d_in[5];
    float* out = (float*)d_out;

    // workspace: s (16*512 f32 = 32 KB) then wd (16*512*512 f16 = 8 MB)
    float*    s  = (float*)d_ws;
    _Float16* wd = (_Float16*)((char*)d_ws + (size_t)kB * kIN * sizeof(float));

    style_kernel<<<dim3(kB), dim3(kWF), 0, stream>>>(w, A_w, A_b, s);
    demod_kernel<<<dim3(kOUT, kB), dim3(128), 0, stream>>>(weight, s, wd);
    mod_gemm_kernel<<<dim3(kN / BM, kOUT / BN, kB), dim3(256), 0, stream>>>(x, wd, bias, out);
}
